// MemoryGraph_25082609008979
// MI455X (gfx1250) — matrix operand order repaired and checked by reading, not compiled, not run on hardware
//
#include <hip/hip_runtime.h>
#include <math.h>

typedef __attribute__((ext_vector_type(16))) _Float16 v16h;
typedef __attribute__((ext_vector_type(8)))  _Float16 v8h;
typedef __attribute__((ext_vector_type(4)))  _Float16 v4h;
typedef __attribute__((ext_vector_type(8)))  float    v8f;
typedef _Float16 half_t;

#define N_CELLS 128
#define C_N     256
#define K_NBR   16
#define D_N     64
#define HS      128
#define HM      128
#define HMOD    64
#define HEADS   4
#define DH      16

#define CT      8                 // neurons per block
#define EDGES   (CT * K_NBR)      // 128 edge rows per block
#define BLOCK   256
#define WAVES   8

// ---------------- fast tanh: gfx1250 v_tanh_f32 TRANS op (co-executes with WMMA) ----
__device__ inline float fast_tanh(float x) {
#if __has_builtin(__builtin_amdgcn_tanh_f32)
  return __builtin_amdgcn_tanh_f32(x);
#elif __has_builtin(__builtin_amdgcn_tanhf)
  return __builtin_amdgcn_tanhf(x);
#else
  return tanhf(x);
#endif
}

__device__ inline float gelu_tanh(float x) {
  const float x3 = x * x * x;
  return 0.5f * x * (1.0f + fast_tanh(0.7978845608028654f * (x + 0.044715f * x3)));
}

// ---------------- WMMA fragment helpers (wave32 layouts per CDNA5 ISA) ----------------

// A-matrix 16x32 f16: lanes 0-15 -> M=lane,    K = k0 + {0..7, 16..23};
//                     lanes 16-31 -> M=lane-16, K = k0 + {8..15, 24..31}
// Row-major source, 16-byte aligned chunks -> 2x ds_load_b128 per fragment.
__device__ inline v16h load_a16(const half_t* __restrict__ base, int stride, int lane, int k0) {
  const int m  = lane & 15;
  const int kb = k0 + ((lane >> 4) << 3);
  const half_t* row = base + m * stride + kb;
  const v8h lo = *(const v8h*)(row);
  const v8h hi = *(const v8h*)(row + 16);
  v16h a;
#pragma unroll
  for (int i = 0; i < 8; ++i) { a[i] = lo[i]; a[8 + i] = hi[i]; }
  return a;
}

// B fragments pre-swizzled in LDS: fragment (ks, nt) = 512 halves laid out [lane][i],
// element (lane, i) = W[K = ks*32 + (lane>>4)*8 + (i&7) + (i>>3)*16][N = nt*16 + (lane&15)].
// Per-lane read = 32 contiguous bytes -> 2x ds_load_b128.
__device__ inline v16h load_b_swz(const half_t* __restrict__ buf, int NT, int lane, int ks, int nt) {
  const half_t* p = buf + (((ks * NT + nt) << 9) + (lane << 4));
  return *(const v16h*)p;
}

__device__ inline v8f wmma_f16(v16h a, v16h b, v8f c) {
  return __builtin_amdgcn_wmma_f32_16x16x32_f16(false, a, false, b, (short)0, c, false, false);
}

// Cooperative staging of a row-major [Ntot][Ktot] f32 weight into the swizzled B layout.
__device__ inline void stage_b_swz(half_t* __restrict__ dst, const float* __restrict__ w,
                                   int Ktot, int Ntot, int tid) {
  const int NT = Ntot >> 4;
  for (int idx = tid; idx < Ktot * Ntot; idx += BLOCK) {
    const int frag = idx >> 9;
    const int ln   = (idx >> 4) & 31;
    const int ii   = idx & 15;
    const int ks   = frag / NT;
    const int nt   = frag - ks * NT;
    const int k    = ks * 32 + ((ln >> 4) << 3) + (ii & 7) + ((ii >> 3) << 4);
    const int nn   = nt * 16 + (ln & 15);
    dst[idx] = (half_t)w[nn * Ktot + k];
  }
}

// ---------------- Kernel A: per-cell modulation gate ----------------
__global__ __launch_bounds__(64) void mg_gate_kernel(
    const float* __restrict__ h,
    const float* __restrict__ mod_w1, const float* __restrict__ mod_b1,
    const float* __restrict__ mod_w2, const float* __restrict__ mod_b2,
    float* __restrict__ gate) {
  const int n = blockIdx.x;
  const int t = threadIdx.x;  // 64 threads: one per D / HMOD element
  __shared__ float pooled[D_N];
  __shared__ float mh[HMOD];

  const float* hn = h + (size_t)n * C_N * D_N;
  float s = 0.0f;
  for (int c = 0; c < C_N; ++c) s += hn[c * D_N + t];
  pooled[t] = s * (1.0f / (float)C_N);
  __syncthreads();

  const float* w1 = mod_w1 + (size_t)n * D_N * HMOD;
  float a = mod_b1[n * HMOD + t];
  for (int d = 0; d < D_N; ++d) a += pooled[d] * w1[d * HMOD + t];
  mh[t] = fast_tanh(a);
  __syncthreads();

  const float* w2 = mod_w2 + (size_t)n * HMOD * D_N;
  float o = mod_b2[n * D_N + t];
  for (int q = 0; q < HMOD; ++q) o += mh[q] * w2[q * D_N + t];
  gate[n * D_N + t] = fast_tanh(o);
}

// ---------------- Kernel B: fused attention + message MLP + state MLP ----------------
__global__ __launch_bounds__(BLOCK) void mg_main_kernel(
    const float* __restrict__ h, const float* __restrict__ neuron_id,
    const float* __restrict__ neuron_key,
    const float* __restrict__ state_w1, const float* __restrict__ state_b1,
    const float* __restrict__ state_gs1, const float* __restrict__ state_gb1,
    const float* __restrict__ state_w2, const float* __restrict__ state_b2,
    const float* __restrict__ state_gs2, const float* __restrict__ state_gb2,
    const float* __restrict__ msg_w1, const float* __restrict__ msg_b1,
    const float* __restrict__ msg_gs1, const float* __restrict__ msg_gb1,
    const float* __restrict__ msg_w2, const float* __restrict__ msg_b2,
    const float* __restrict__ msg_gs2, const float* __restrict__ msg_gb2,
    const int* __restrict__ conn_idx, const int* __restrict__ cell_to_group,
    const float* __restrict__ gate, float* __restrict__ out) {

  const int n   = blockIdx.x;        // cell
  const int c0  = blockIdx.y * CT;   // first neuron of this tile
  const int tid = threadIdx.x;
  const int lane = tid & 31;
  const int wave = tid >> 5;
  const int grp = cell_to_group[n];

  __shared__ half_t Xbuf[EDGES][2 * D_N];       // 32 KB: edge inputs, later s_in (rows 0..15)
  __shared__ half_t WbufS[2 * D_N * HM];        // 32 KB: swizzled B frags, msg_w1 then state_w1
  __shared__ half_t W2bufS[HM * D_N];           // 16 KB: swizzled B frags, msg_w2 then state_w2
  __shared__ half_t M1s[WAVES][16][HM];         // 32 KB: per-wave m1 tile; M1s[0] reused as s1
  __shared__ float attnS[CT][K_NBR][HEADS];
  __shared__ float aggS[CT][D_N];
  __shared__ float qS[CT][D_N];
  __shared__ int   connS[CT][K_NBR];
  __shared__ float p_b1[HM], p_gs1[HM], p_gb1[HM];
  __shared__ float p_b2[D_N], p_gs2[D_N], p_gb2[D_N];

  const float* hn  = h         + (size_t)n * C_N * D_N;
  const float* idn = neuron_id + (size_t)n * C_N * D_N;

  // ---- cooperative staging: indices, queries, params, msg weights (swizzled) ----
  if (tid < CT * K_NBR)
    connS[tid >> 4][tid & 15] = conn_idx[((size_t)n * C_N + c0 + (tid >> 4)) * K_NBR + (tid & 15)];
  for (int i = tid; i < CT * D_N; i += BLOCK)
    qS[i >> 6][i & 63] = neuron_key[((size_t)n * C_N + c0 + (i >> 6)) * D_N + (i & 63)];
  for (int i = tid; i < HM; i += BLOCK) {
    p_b1[i] = msg_b1[i]; p_gs1[i] = msg_gs1[grp * HM + i]; p_gb1[i] = msg_gb1[grp * HM + i];
  }
  for (int i = tid; i < D_N; i += BLOCK) {
    p_b2[i] = msg_b2[i]; p_gs2[i] = msg_gs2[grp * D_N + i]; p_gb2[i] = msg_gb2[grp * D_N + i];
  }
  stage_b_swz(WbufS,  msg_w1, 2 * D_N, HM,  tid);
  stage_b_swz(W2bufS, msg_w2, HM,      D_N, tid);
  __syncthreads();

  // ---- gather edge inputs x = [nbr | h] into Xbuf (f16, packed 8-byte stores) ----
  {
    const int e = tid >> 1, p = tid & 1;        // 128 edges, half-row each
    const int cl = e >> 4, k = e & 15;
    const float* src = (p == 0) ? (hn + (size_t)connS[cl][k] * D_N)
                                : (hn + (size_t)(c0 + cl) * D_N);
    v4h* dst4 = (v4h*)&Xbuf[e][p * D_N];
#pragma unroll
    for (int j = 0; j < D_N; j += 4) {
      const float4 v = *(const float4*)(src + j);
      v4h o; o[0] = (half_t)v.x; o[1] = (half_t)v.y; o[2] = (half_t)v.z; o[3] = (half_t)v.w;
      dst4[j >> 2] = o;
    }
  }

  // ---- attention logits (tiny: keep f32 VALU) ----
  for (int it = tid; it < CT * K_NBR * HEADS; it += BLOCK) {
    const int hd = it & 3, k = (it >> 2) & 15, cl = it >> 6;
    const float* kv = idn + (size_t)connS[cl][k] * D_N + hd * DH;
    const float* qv = &qS[cl][hd * DH];
    float s = 0.0f;
#pragma unroll
    for (int j = 0; j < DH; ++j) s += qv[j] * kv[j];
    attnS[cl][k][hd] = s * 0.25f;  // 1/sqrt(dh=16)
  }
  __syncthreads();
  if (tid < CT * HEADS) {
    const int hd = tid & 3, cl = tid >> 2;
    float mx = -1e30f;
    for (int k = 0; k < K_NBR; ++k) mx = fmaxf(mx, attnS[cl][k][hd]);
    float ex[K_NBR], ssum = 0.0f;
    for (int k = 0; k < K_NBR; ++k) { ex[k] = __expf(attnS[cl][k][hd] - mx); ssum += ex[k]; }
    const float inv = 1.0f / ssum;
    for (int k = 0; k < K_NBR; ++k) attnS[cl][k][hd] = ex[k] * inv;
  }
  __syncthreads();

  // ---- Stage 1+2: per-edge message MLP; one wave == one neuron's 16-edge M-tile ----
  {
    const int cl = wave;
    const half_t* Arows = &Xbuf[cl * K_NBR][0];
    const int mrow = lane & 15, hi = lane >> 4;

    // m1 = gelu(X @ W1^T): all B fragments of a K-step land in distinct registers,
    // so the 9 ds_load_b128 pairs issue ahead of the 8-WMMA burst (no dscnt-0 stall,
    // no WMMA->load WAR NOPs).
    v8f acc[8] = {};
#pragma unroll
    for (int ks = 0; ks < 4; ++ks) {
      const v16h a = load_a16(Arows, 2 * D_N, lane, ks * 32);
      v16h b[8];
#pragma unroll
      for (int nt = 0; nt < 8; ++nt) b[nt] = load_b_swz(WbufS, 8, lane, ks, nt);
#pragma unroll
      for (int nt = 0; nt < 8; ++nt) acc[nt] = wmma_f16(a, b[nt], acc[nt]);
    }
    // m1 epilogue: bias + gelu + group gain -> f16 scratch (A-layout source for stage 2)
#pragma unroll
    for (int nt = 0; nt < 8; ++nt) {
#pragma unroll
      for (int r = 0; r < 8; ++r) {
        const int m = r + hi * 8;
        const int col = nt * 16 + mrow;
        float v = acc[nt][r] + p_b1[col];
        v = gelu_tanh(v);
        v = v * p_gs1[col] + p_gb1[col];
        M1s[wave][m][col] = (half_t)v;
      }
    }

    v8f acc2[4] = {};
#pragma unroll
    for (int ks = 0; ks < 4; ++ks) {
      const v16h a = load_a16(&M1s[wave][0][0], HM, lane, ks * 32);
      v16h b[4];
#pragma unroll
      for (int nt = 0; nt < 4; ++nt) b[nt] = load_b_swz(W2bufS, 4, lane, ks, nt);
#pragma unroll
      for (int nt = 0; nt < 4; ++nt) acc2[nt] = wmma_f16(a, b[nt], acc2[nt]);
    }
    // m2 epilogue fused with attention-weighted aggregation over the 16 edge rows.
    // N-tile index nt == head (dh = 16).
#pragma unroll
    for (int nt = 0; nt < 4; ++nt) {
      float part = 0.0f;
#pragma unroll
      for (int r = 0; r < 8; ++r) {
        const int kk = r + hi * 8;           // neighbor index
        const int d  = nt * 16 + mrow;
        float v = acc2[nt][r] + p_b2[d];
        v = v * p_gs2[d] + p_gb2[d];
        part += attnS[cl][kk][nt] * v;
      }
      part += __shfl_xor(part, 16, 32);      // combine the two half-wave row sets
      if (hi == 0) aggS[cl][nt * 16 + mrow] = part;
    }
  }
  __syncthreads();

  // ---- swap LDS buffers to state-MLP weights/params; build s_in = [h | agg] ----
  for (int i = tid; i < HS; i += BLOCK) {
    p_b1[i] = state_b1[i]; p_gs1[i] = state_gs1[grp * HS + i]; p_gb1[i] = state_gb1[grp * HS + i];
  }
  for (int i = tid; i < D_N; i += BLOCK) {
    p_b2[i] = state_b2[i]; p_gs2[i] = state_gs2[grp * D_N + i]; p_gb2[i] = state_gb2[grp * D_N + i];
  }
  stage_b_swz(WbufS,  state_w1, 2 * D_N, HS,  tid);
  stage_b_swz(W2bufS, state_w2, HS,      D_N, tid);
  for (int i = tid; i < 16 * 2 * D_N; i += BLOCK) {   // 16-row tile, rows >= CT zero-padded
    const int r = i >> 7, col = i & 127;
    float v = 0.0f;
    if (r < CT) v = (col < D_N) ? hn[(size_t)(c0 + r) * D_N + col] : aggS[r][col - D_N];
    Xbuf[r][col] = (half_t)v;
  }
  __syncthreads();

  // ---- Stage 3a: s1 = gelu(s_in @ state_w1^T) * gs + gb ; one N-tile per wave ----
  {
    const int nt = wave;
    // Preload all 4 K-steps (A and B) before the serial accumulator chain.
    v16h a[4], b[4];
#pragma unroll
    for (int ks = 0; ks < 4; ++ks) {
      a[ks] = load_a16(&Xbuf[0][0], 2 * D_N, lane, ks * 32);
      b[ks] = load_b_swz(WbufS, 8, lane, ks, nt);
    }
    v8f acc = {};
#pragma unroll
    for (int ks = 0; ks < 4; ++ks) acc = wmma_f16(a[ks], b[ks], acc);

    const int mrow = lane & 15, hi = lane >> 4;
#pragma unroll
    for (int r = 0; r < 8; ++r) {
      const int m = r + hi * 8;
      const int col = nt * 16 + mrow;
      float v = acc[r] + p_b1[col];
      v = gelu_tanh(v);
      v = v * p_gs1[col] + p_gb1[col];
      M1s[0][m][col] = (half_t)v;              // s1 buffer (columns disjoint per wave)
    }
  }
  __syncthreads();

  // ---- Stage 3b: delta = s1 @ state_w2^T ; fuse gain + gate + residual store ----
  if (wave < 4) {
    const int nt = wave;
    v16h a[4], b[4];
#pragma unroll
    for (int ks = 0; ks < 4; ++ks) {
      a[ks] = load_a16(&M1s[0][0][0], HS, lane, ks * 32);
      b[ks] = load_b_swz(W2bufS, 4, lane, ks, nt);
    }
    v8f acc = {};
#pragma unroll
    for (int ks = 0; ks < 4; ++ks) acc = wmma_f16(a[ks], b[ks], acc);

    const int mrow = lane & 15, hi = lane >> 4;
#pragma unroll
    for (int r = 0; r < 8; ++r) {
      const int m = r + hi * 8;                // neuron row; valid if < CT
      if (m < CT) {
        const int d = nt * 16 + mrow;
        float v = acc[r] + p_b2[d];
        v = v * p_gs2[d] + p_gb2[d];
        const size_t off = ((size_t)n * C_N + c0 + m) * D_N + d;
        out[off] = h[off] + gate[n * D_N + d] * v;
      }
    }
  }
}

// ---------------- launch ----------------
extern "C" void kernel_launch(void* const* d_in, const int* in_sizes, int n_in,
                              void* d_out, int out_size, void* d_ws, size_t ws_size,
                              hipStream_t stream) {
  const float* h          = (const float*)d_in[0];
  const float* neuron_id  = (const float*)d_in[1];
  const float* neuron_key = (const float*)d_in[2];
  const float* state_w1   = (const float*)d_in[3];
  const float* state_b1   = (const float*)d_in[4];
  const float* state_gs1  = (const float*)d_in[5];
  const float* state_gb1  = (const float*)d_in[6];
  const float* state_w2   = (const float*)d_in[7];
  const float* state_b2   = (const float*)d_in[8];
  const float* state_gs2  = (const float*)d_in[9];
  const float* state_gb2  = (const float*)d_in[10];
  const float* msg_w1     = (const float*)d_in[11];
  const float* msg_b1     = (const float*)d_in[12];
  const float* msg_gs1    = (const float*)d_in[13];
  const float* msg_gb1    = (const float*)d_in[14];
  const float* msg_w2     = (const float*)d_in[15];
  const float* msg_b2     = (const float*)d_in[16];
  const float* msg_gs2    = (const float*)d_in[17];
  const float* msg_gb2    = (const float*)d_in[18];
  const float* mod_w1     = (const float*)d_in[19];
  const float* mod_b1     = (const float*)d_in[20];
  const float* mod_w2     = (const float*)d_in[21];
  const float* mod_b2     = (const float*)d_in[22];
  const int*   conn_idx      = (const int*)d_in[23];
  const int*   cell_to_group = (const int*)d_in[24];

  float* gate = (float*)d_ws;            // [N_CELLS * D_N] floats
  float* out  = (float*)d_out;

  mg_gate_kernel<<<dim3(N_CELLS), dim3(64), 0, stream>>>(h, mod_w1, mod_b1, mod_w2, mod_b2, gate);

  mg_main_kernel<<<dim3(N_CELLS, C_N / CT), dim3(BLOCK), 0, stream>>>(
      h, neuron_id, neuron_key,
      state_w1, state_b1, state_gs1, state_gb1, state_w2, state_b2, state_gs2, state_gb2,
      msg_w1, msg_b1, msg_gs1, msg_gb1, msg_w2, msg_b2, msg_gs2, msg_gb2,
      conn_idx, cell_to_group, gate, out);
}
